// xFormersWithBuffer_31061203484870
// MI455X (gfx1250) — compile-verified, matrix operand order checked by reading, not run on hardware
//
#include <hip/hip_runtime.h>

typedef __attribute__((ext_vector_type(16))) __bf16 v16bf;
typedef __attribute__((ext_vector_type(2)))  __bf16 v2bf;
typedef __attribute__((ext_vector_type(8)))  float  v8f;
typedef __attribute__((ext_vector_type(2)))  float  v2f;

#define N_SEQS    8
#define Q_LEN     256
#define KV_LEN    2048
#define NUM_HEADS 16
#define HEAD_DIM  128
#define MODEL_DIM (NUM_HEADS * HEAD_DIM)   // 2048 floats per token row
#define CTX_OFF   (KV_LEN - Q_LEN)         // 1792 (chunk-aligned: 28*64)
#define CHUNK     64
#define SCALING   0.08838834764831845f     // 1/sqrt(128), folded into Q
#define NEG_BIG   (-3.0e38f)

// Native packed fp32 -> bf16 (RNE): lets the backend use v_cvt_pk_bf16_f32.
__device__ __forceinline__ unsigned pack2(float a, float b) {
  v2f s; s.x = a; s.y = b;
  v2bf t = __builtin_convertvector(s, v2bf);
  union { v2bf b; unsigned u; } u;
  u.b = t;
  return u.u;
}
__device__ __forceinline__ unsigned short f2bf(float f) {
  union { __bf16 b; unsigned short u; } x;
  x.b = (__bf16)f;
  return x.u;
}

union FragU { v16bf v; uint4 q[2]; };

// 16x32 A-fragment / 32x16 B-fragment (bf16) from LDS.
// lane l: row = row_base + (l&15); elements kb + (l>>4)*8 .. +7 and +16 .. +23
__device__ __forceinline__ v16bf ld_frag_lds(const unsigned short* base,
                                             int row_base, int stride, int kb) {
  const int l = (int)(threadIdx.x & 31);
  const unsigned short* p = base + (row_base + (l & 15)) * stride + kb + ((l >> 4) << 3);
  FragU u;
  u.q[0] = *(const uint4*)(p);
  u.q[1] = *(const uint4*)(p + 16);
  return u.v;
}

// Q A-fragment from global fp32, pre-scaled by 1/sqrt(d).
__device__ __forceinline__ v16bf q_frag_global(const float* rowp, int kb) {
  const int l = (int)(threadIdx.x & 31);
  const float* p = rowp + kb + ((l >> 4) << 3);
  float4 a0 = *(const float4*)(p);
  float4 a1 = *(const float4*)(p + 4);
  float4 b0 = *(const float4*)(p + 16);
  float4 b1 = *(const float4*)(p + 20);
  FragU u;
  u.q[0] = make_uint4(pack2(a0.x * SCALING, a0.y * SCALING),
                      pack2(a0.z * SCALING, a0.w * SCALING),
                      pack2(a1.x * SCALING, a1.y * SCALING),
                      pack2(a1.z * SCALING, a1.w * SCALING));
  u.q[1] = make_uint4(pack2(b0.x * SCALING, b0.y * SCALING),
                      pack2(b0.z * SCALING, b0.w * SCALING),
                      pack2(b1.x * SCALING, b1.y * SCALING),
                      pack2(b1.z * SCALING, b1.w * SCALING));
  return u.v;
}

struct RowRegs { float4 a[4]; };   // 16 floats = one kv row slice [cs .. cs+15]

__device__ __forceinline__ RowRegs ld_row(const float* src, int cs) {
  RowRegs r;
#pragma unroll
  for (int i = 0; i < 4; ++i) r.a[i] = *(const float4*)(src + cs + i * 4);
  return r;
}

__device__ __forceinline__ void st_k(unsigned short* KTb, int row, int cs,
                                     const RowRegs& r) {
  uint4 s0 = make_uint4(pack2(r.a[0].x, r.a[0].y), pack2(r.a[0].z, r.a[0].w),
                        pack2(r.a[1].x, r.a[1].y), pack2(r.a[1].z, r.a[1].w));
  uint4 s1 = make_uint4(pack2(r.a[2].x, r.a[2].y), pack2(r.a[2].z, r.a[2].w),
                        pack2(r.a[3].x, r.a[3].y), pack2(r.a[3].z, r.a[3].w));
  *(uint4*)&KTb[row * HEAD_DIM + cs] = s0;
  *(uint4*)&KTb[row * HEAD_DIM + cs + 8] = s1;
}

// V transposed: element (d=cs+i, kv=p2 / p2+1) packed into one dword store.
__device__ __forceinline__ void st_vt(unsigned short* VTb, int p2, int cs,
                                      const RowRegs& r0, const RowRegs& r1) {
#pragma unroll
  for (int i = 0; i < 4; ++i) {
    const float4 x = r0.a[i], y = r1.a[i];
    *(unsigned*)&VTb[(cs + i * 4 + 0) * CHUNK + p2] = pack2(x.x, y.x);
    *(unsigned*)&VTb[(cs + i * 4 + 1) * CHUNK + p2] = pack2(x.y, y.y);
    *(unsigned*)&VTb[(cs + i * 4 + 2) * CHUNK + p2] = pack2(x.z, y.z);
    *(unsigned*)&VTb[(cs + i * 4 + 3) * CHUNK + p2] = pack2(x.w, y.w);
  }
}

// Gather sources for kv rows (p2, p2+1) of chunk c (chunks never straddle
// CTX_OFF since 1792 = 28*64).
__device__ __forceinline__ void resolve_src(int c, int p2, int seq, int h,
                                            const float* __restrict__ k,
                                            const float* __restrict__ v,
                                            const float* __restrict__ kc,
                                            const float* __restrict__ vc,
                                            const int*  __restrict__ ctx,
                                            const float*& k0, const float*& v0,
                                            const float*& k1, const float*& v1) {
  const int j0 = c * CHUNK + p2;
  if (j0 >= CTX_OFF) {
    size_t r0 = (size_t)(seq * Q_LEN + (j0 - CTX_OFF)) * MODEL_DIM + h * HEAD_DIM;
    k0 = k + r0;             v0 = v + r0;
    k1 = k + r0 + MODEL_DIM; v1 = v + r0 + MODEL_DIM;
  } else {
    int2 ss = *(const int2*)(ctx + seq * KV_LEN + j0);
    size_t r0 = (size_t)ss.x * MODEL_DIM + h * HEAD_DIM;
    size_t r1 = (size_t)ss.y * MODEL_DIM + h * HEAD_DIM;
    k0 = kc + r0; v0 = vc + r0;
    k1 = kc + r1; v1 = vc + r1;
  }
}

__global__ __launch_bounds__(256)
void fill_attn_kernel(const float* __restrict__ q,  const float* __restrict__ k,
                      const float* __restrict__ v,  const float* __restrict__ k_cache,
                      const float* __restrict__ v_cache,
                      const int*   __restrict__ context_idx,
                      float* __restrict__ out) {
  __shared__ __align__(16) unsigned short KT[2][CHUNK * HEAD_DIM];   // [kv][d]
  __shared__ __align__(16) unsigned short VT[2][HEAD_DIM * CHUNK];   // [d][kv]
  __shared__ __align__(16) unsigned short Pw[8][16 * CHUNK];         // per-wave P

  const int sh   = (int)blockIdx.x;        // seq*16 + head
  const int seq  = sh >> 4;
  const int h    = sh & 15;
  const int wg_q = (int)blockIdx.y * 128;
  const int wave = (int)(threadIdx.x >> 5);
  const int lane = (int)(threadIdx.x & 31);
  const int qbase = wg_q + wave * 16;

  // ---- Q fragments (resident in VGPRs, pre-scaled by 1/sqrt(d)) ----
  const float* qrow = q + (size_t)(seq * Q_LEN + qbase + (lane & 15)) * MODEL_DIM
                        + h * HEAD_DIM;
  v16bf qf[4];
#pragma unroll
  for (int kc2 = 0; kc2 < 4; ++kc2) qf[kc2] = q_frag_global(qrow, kc2 * 32);

  // constant all-ones B fragment: rowsum(P) = P x ones via WMMA
  FragU uo;
  uo.q[0] = make_uint4(0x3F803F80u, 0x3F803F80u, 0x3F803F80u, 0x3F803F80u);
  uo.q[1] = uo.q[0];
  const v16bf ones = uo.v;

  // ---- flash-attention state ----
  v8f O[8] = {};
  v8f lacc = {};                       // softmax denominator, C-layout like O
  float mrun[8];
#pragma unroll
  for (int g = 0; g < 8; ++g) mrun[g] = NEG_BIG;

  const int kv_end_raw = wg_q + 127 + CTX_OFF + 1;                 // 1920 or 2048
  const int kv_end     = kv_end_raw < KV_LEN ? kv_end_raw : KV_LEN;
  const int nchunks    = (kv_end + CHUNK - 1) / CHUNK;
  const int jlimit     = qbase + 15 + CTX_OFF;

  // loader mapping: 256 thr -> 32 row-pairs x 8 column-slices of 16 dims
  const int p2 = (int)(threadIdx.x >> 3) * 2;        // kv rows p2, p2+1
  const int cs = (int)(threadIdx.x & 7) * 16;        // d columns cs..cs+15

  // ---- preload chunk 0 into buffer 0 ----
  {
    const float *k0, *v0, *k1, *v1;
    resolve_src(0, p2, seq, h, k, v, k_cache, v_cache, context_idx, k0, v0, k1, v1);
    RowRegs kr0 = ld_row(k0, cs), kr1 = ld_row(k1, cs);
    RowRegs vr0 = ld_row(v0, cs), vr1 = ld_row(v1, cs);
    st_k(KT[0], p2, cs, kr0);  st_k(KT[0], p2 + 1, cs, kr1);
    st_vt(VT[0], p2, cs, vr0, vr1);
  }

  for (int c = 0; c < nchunks; ++c) {
    // ---- stage chunk c+1 into registers (loads fly during compute of c) ----
    RowRegs kr0, kr1, vr0, vr1;
    const bool have_next = (c + 1 < nchunks);
    if (have_next) {
      const float *k0, *v0, *k1, *v1;
      resolve_src(c + 1, p2, seq, h, k, v, k_cache, v_cache, context_idx,
                  k0, v0, k1, v1);
      kr0 = ld_row(k0, cs);  kr1 = ld_row(k1, cs);
      vr0 = ld_row(v0, cs);  vr1 = ld_row(v1, cs);
    }
    // L2 warm-up two chunks ahead
    if (c + 2 < nchunks) {
      const float *pk0, *pv0, *pk1, *pv1;
      resolve_src(c + 2, p2, seq, h, k, v, k_cache, v_cache, context_idx,
                  pk0, pv0, pk1, pv1);
      __builtin_prefetch(pk0 + cs, 0, 0);
      __builtin_prefetch(pv0 + cs, 0, 0);
    }

    __syncthreads();   // tile for chunk c (buffer c&1) ready across all waves

    const unsigned short* Ktc = KT[c & 1];
    const unsigned short* Vtc = VT[c & 1];

    if (c * CHUNK <= jlimit) {
      // ---- S = Qs K^T ----
      v8f S[4];
#pragma unroll
      for (int nt = 0; nt < 4; ++nt) {
        v8f acc = {};
#pragma unroll
        for (int kc2 = 0; kc2 < 4; ++kc2) {
          v16bf bfrag = ld_frag_lds(Ktc, nt * 16, HEAD_DIM, kc2 * 32);
          acc = __builtin_amdgcn_wmma_f32_16x16x32_bf16(false, qf[kc2], false, bfrag,
                                                        (short)0, acc, false, false);
        }
        S[nt] = acc;
      }

      // ---- bottom-right causal mask: only the boundary chunk needs it ----
      if (c * CHUNK + CHUNK - 1 > qbase + CTX_OFF) {
#pragma unroll
        for (int g = 0; g < 8; ++g) {
          const int mglob = qbase + g + ((lane >> 4) << 3);
#pragma unroll
          for (int nt = 0; nt < 4; ++nt) {
            int jg2 = c * CHUNK + nt * 16 + (lane & 15);
            S[nt][g] = (jg2 <= mglob + CTX_OFF) ? S[nt][g] : NEG_BIG;
          }
        }
      }

      // ---- per-row chunk max (cross-lane reduce over 16 lanes) ----
      float chmax[8];
#pragma unroll
      for (int g = 0; g < 8; ++g) {
        float mx = fmaxf(fmaxf(S[0][g], S[1][g]), fmaxf(S[2][g], S[3][g]));
#pragma unroll
        for (int sft = 8; sft >= 1; sft >>= 1)
          mx = fmaxf(mx, __shfl_xor(mx, sft, 16));
        chmax[g] = mx;
      }

      // ---- online softmax; stage P (bf16) in per-wave LDS ----
      unsigned short* pw = Pw[wave];
#pragma unroll
      for (int g = 0; g < 8; ++g) {
        const float m2    = fmaxf(mrun[g], chmax[g]);
        const float scale = __expf(mrun[g] - m2);
        mrun[g] = m2;
        const int row = g + ((lane >> 4) << 3);
#pragma unroll
        for (int nt = 0; nt < 4; ++nt)
          pw[row * CHUNK + nt * 16 + (lane & 15)] = f2bf(__expf(S[nt][g] - m2));
        lacc[g] *= scale;
#pragma unroll
        for (int dt = 0; dt < 8; ++dt) O[dt][g] *= scale;
      }
      asm volatile("" ::: "memory");  // per-wave LDS stores precede reloads (LDS in-order)

      // ---- O += P V ; denominator += P x ones (matrix-engine rowsum) ----
      v16bf pf0 = ld_frag_lds(pw, 0, CHUNK, 0);
      v16bf pf1 = ld_frag_lds(pw, 0, CHUNK, 32);
      lacc = __builtin_amdgcn_wmma_f32_16x16x32_bf16(false, pf0, false, ones,
                                                     (short)0, lacc, false, false);
      lacc = __builtin_amdgcn_wmma_f32_16x16x32_bf16(false, pf1, false, ones,
                                                     (short)0, lacc, false, false);
#pragma unroll
      for (int dt = 0; dt < 8; ++dt) {
        v16bf vf0 = ld_frag_lds(Vtc, dt * 16, CHUNK, 0);
        v16bf vf1 = ld_frag_lds(Vtc, dt * 16, CHUNK, 32);
        O[dt] = __builtin_amdgcn_wmma_f32_16x16x32_bf16(false, pf0, false, vf0,
                                                        (short)0, O[dt], false, false);
        O[dt] = __builtin_amdgcn_wmma_f32_16x16x32_bf16(false, pf1, false, vf1,
                                                        (short)0, O[dt], false, false);
      }
    }

    // ---- drain staged registers into the other buffer (no race: buffer
    // (c+1)&1 was last read before the barrier above) ----
    if (have_next) {
      unsigned short* ktn = KT[(c + 1) & 1];
      unsigned short* vtn = VT[(c + 1) & 1];
      st_k(ktn, p2, cs, kr0);  st_k(ktn, p2 + 1, cs, kr1);
      st_vt(vtn, p2, cs, vr0, vr1);
    }
  }

  // ---- normalize and store fp32 output [2048, 2048] ----
  const int orow = seq * Q_LEN + qbase;
#pragma unroll
  for (int g = 0; g < 8; ++g) {
    const float inv = 1.0f / lacc[g];
    const int m = g + ((lane >> 4) << 3);
    float* op = out + (size_t)(orow + m) * MODEL_DIM + h * HEAD_DIM + (lane & 15);
#pragma unroll
    for (int dt = 0; dt < 8; ++dt)
      op[dt * 16] = O[dt][g] * inv;
  }
}

extern "C" void kernel_launch(void* const* d_in, const int* in_sizes, int n_in,
                              void* d_out, int out_size, void* d_ws, size_t ws_size,
                              hipStream_t stream) {
  const float* q       = (const float*)d_in[0];
  const float* k       = (const float*)d_in[1];
  const float* v       = (const float*)d_in[2];
  const float* k_cache = (const float*)d_in[3];
  const float* v_cache = (const float*)d_in[4];
  // d_in[5] = allocated_idx: unused — equals ctx[:, -Q_LEN:] by construction,
  // so the cache scatter is folded into the gather (j >= 1792 -> new k/v rows).
  const int* context_idx = (const int*)d_in[6];

  dim3 grid(N_SEQS * NUM_HEADS, Q_LEN / 128);  // 128 x 2 workgroups
  fill_attn_kernel<<<grid, 256, 0, stream>>>(q, k, v, k_cache, v_cache,
                                             context_idx, (float*)d_out);
}
